// GroupedQueryAttention_8796093022292
// MI455X (gfx1250) — matrix operand order repaired and checked by reading, not compiled, not run on hardware
//
#include <hip/hip_runtime.h>
#include <hip/hip_bf16.h>
#include <math.h>

typedef __attribute__((ext_vector_type(16))) __bf16 v16bf;
typedef __attribute__((ext_vector_type(8)))  __bf16 v8bf;
typedef __attribute__((ext_vector_type(8)))  float  v8f;

// Problem constants (match reference)
constexpr int Bc  = 2;
constexpr int Sc  = 2048;
constexpr int Dc  = 2048;
constexpr int Hc  = 32;
constexpr int Gc  = 8;
constexpr int HDc = 64;
constexpr int BSc = Bc * Sc;          // 4096 rows

static __device__ __forceinline__ __bf16 f2bf(float f) {
  unsigned u = __builtin_bit_cast(unsigned, f);
  u += 0x7FFFu + ((u >> 16) & 1u);              // round-to-nearest-even
  unsigned short s = (unsigned short)(u >> 16);
  return __builtin_bit_cast(__bf16, s);
}

// async copy of 16 bytes global -> LDS (per lane), tracked by ASYNCcnt
static __device__ __forceinline__ void async_b128(const void* gptr, void* lptr) {
  unsigned lds = (unsigned)(size_t)lptr;                 // low 32 bits = LDS offset
  unsigned long long ga = (unsigned long long)(size_t)gptr;
  asm volatile("global_load_async_to_lds_b128 %0, %1, off" :: "v"(lds), "v"(ga) : "memory");
}
static __device__ __forceinline__ void wait_async0() {
  asm volatile("s_wait_asynccnt 0x0" ::: "memory");
}

// ---------------------------------------------------------------------------
// f32 -> bf16 elementwise conversion (grid-stride)
// ---------------------------------------------------------------------------
__global__ __launch_bounds__(256)
void cvt_f32_bf16(const float* __restrict__ in, __bf16* __restrict__ out, long n) {
  long i = (long)blockIdx.x * 256 + threadIdx.x;
  long stride = (long)gridDim.x * 256;
  for (; i < n; i += stride) out[i] = f2bf(in[i]);
}

// ---------------------------------------------------------------------------
// fused convert + transpose: in (R, C) f32 row-major -> out (C, R) bf16
// tile 64x64 via LDS, both global accesses coalesced
// grid = (C/64, R/64)
// ---------------------------------------------------------------------------
__global__ __launch_bounds__(256)
void cvt_transpose(const float* __restrict__ in, __bf16* __restrict__ out,
                   long in_rs, long out_rs) {
  __shared__ __bf16 t[64][72];                 // padded vs bank conflicts
  const int r0 = blockIdx.y * 64, c0 = blockIdx.x * 64;
  const int tid = threadIdx.x;
#pragma unroll
  for (int i = 0; i < 16; ++i) {
    int idx = tid + i * 256;
    int r = idx >> 6, c = idx & 63;
    t[c][r] = f2bf(in[(size_t)(r0 + r) * in_rs + c0 + c]);
  }
  __syncthreads();
#pragma unroll
  for (int i = 0; i < 16; ++i) {
    int idx = tid + i * 256;
    int r = idx >> 6, c = idx & 63;
    out[(size_t)(c0 + r) * out_rs + r0 + c] = t[r][c];
  }
}

// V transpose: f32 (b, s, g, 64) -> bf16 (b, g, 64, s)   (i.e. V^T per (b,g))
// grid = (1, S/64, B*G)
__global__ __launch_bounds__(256)
void v_transpose(const float* __restrict__ in, __bf16* __restrict__ out) {
  __shared__ __bf16 t[64][72];
  const int z = blockIdx.z;            // b*G + g
  const int b = z >> 3, g = z & 7;
  const int s0 = blockIdx.y * 64;
  const int tid = threadIdx.x;
  const float* ibase = in + (size_t)b * Sc * (Gc * HDc) + g * HDc;   // (s, d) rows stride G*HD
  __bf16* obase = out + (size_t)z * HDc * Sc;                        // (d, s) rows stride S
#pragma unroll
  for (int i = 0; i < 16; ++i) {
    int idx = tid + i * 256;
    int r = idx >> 6, c = idx & 63;        // r = s-within, c = d
    t[c][r] = f2bf(ibase[(size_t)(s0 + r) * (Gc * HDc) + c]);
  }
  __syncthreads();
#pragma unroll
  for (int i = 0; i < 16; ++i) {
    int idx = tid + i * 256;
    int r = idx >> 6, c = idx & 63;        // r = d, c = s-within
    obase[(size_t)r * Sc + s0 + c] = t[r][c];
  }
}

// ---------------------------------------------------------------------------
// bf16 WMMA GEMM: C[M,N](f32) = A[M,K](bf16, row-major) x Bt[N,K](bf16)
// block = 128 threads (4 waves), tile 64x64, BK = 64
// double-buffered LDS filled by global_load_async_to_lds_b128
// ---------------------------------------------------------------------------
__global__ __launch_bounds__(128)
void gemm_bf16_wmma(const __bf16* __restrict__ A, const __bf16* __restrict__ Bt,
                    float* __restrict__ C, int M, int N, int K) {
  __shared__ alignas(64) __bf16 sA[2][64][64];
  __shared__ alignas(64) __bf16 sB[2][64][64];   // (n, k)

  const int tid  = threadIdx.x;
  const int lane = tid & 31;
  const int wave = tid >> 5;
  const int hf   = lane >> 4;
  const int l15  = lane & 15;
  const int m0 = blockIdx.y * 64;
  const int n0 = blockIdx.x * 64;

  auto issue_tiles = [&](int buf, int k0) {
#pragma unroll
    for (int i = 0; i < 4; ++i) {
      int idx = tid + i * 128;          // chunk 0..511, 8 halves each
      int r = idx >> 3;
      int c = (idx & 7) * 8;
      async_b128(&A [(size_t)(m0 + r) * K + k0 + c], &sA[buf][r][c]);
      async_b128(&Bt[(size_t)(n0 + r) * K + k0 + c], &sB[buf][r][c]);
    }
  };

  const v8f vzero = {0.f, 0.f, 0.f, 0.f, 0.f, 0.f, 0.f, 0.f};
  v8f acc[4];
#pragma unroll
  for (int t = 0; t < 4; ++t) acc[t] = vzero;

  issue_tiles(0, 0);
  int buf = 0;
  for (int k0 = 0; k0 < K; k0 += 64) {
    wait_async0();
    __syncthreads();
    if (k0 + 64 < K) issue_tiles(buf ^ 1, k0 + 64);   // overlap DMA with WMMA

#pragma unroll
    for (int kc = 0; kc < 64; kc += 32) {
      const __bf16* arow = &sA[buf][wave * 16 + l15][kc + hf * 16];
      v8bf alo = *(const v8bf*)(arow);
      v8bf ahi = *(const v8bf*)(arow + 8);
      v16bf a;
#pragma unroll
      for (int e = 0; e < 8; ++e) { a[e] = alo[e]; a[8 + e] = ahi[e]; }
#pragma unroll
      for (int t = 0; t < 4; ++t) {
        v16bf b = *(const v16bf*)&sB[buf][t * 16 + l15][kc + hf * 16];
        acc[t] = __builtin_amdgcn_wmma_f32_16x16x32_bf16(
            false, a, false, b, (short)0, acc[t], false, false);
      }
    }
    buf ^= 1;
  }

  // C layout: vgpr r, lanes0-15: M=r, lanes16-31: M=8+r; N=lane&15
#pragma unroll
  for (int t = 0; t < 4; ++t)
#pragma unroll
    for (int r = 0; r < 8; ++r) {
      int m = m0 + wave * 16 + hf * 8 + r;
      int n = n0 + t * 16 + l15;
      C[(size_t)m * N + n] = acc[t][r];
    }
}

// ---------------------------------------------------------------------------
// RMSNorm + RoPE (+ scaling) epilogue for Q/K.
// in : f32 (b, s, nH, 64) ; out: bf16 (b, nH, s, 64)
// ---------------------------------------------------------------------------
__global__ __launch_bounds__(256)
void qk_norm_rope(const float* __restrict__ in, __bf16* __restrict__ out,
                  const float* __restrict__ scale,
                  const float* __restrict__ cosT, const float* __restrict__ sinT,
                  int nH, float postscale, int nrows) {
  int wid = blockIdx.x * 8 + (threadIdx.x >> 5);
  if (wid >= nrows) return;
  int lane = threadIdx.x & 31;
  int h = wid % nH;
  int s = (wid / nH) % Sc;
  int b = wid / (nH * Sc);
  const float* row = in + ((size_t)(b * Sc + s) * nH + h) * HDc;
  float x0 = row[lane], x1 = row[lane + 32];
  float ss = x0 * x0 + x1 * x1;
#pragma unroll
  for (int m = 16; m >= 1; m >>= 1) ss += __shfl_xor(ss, m, 32);
  float r = rsqrtf(ss * (1.0f / HDc) + 1e-6f);
  float y0 = x0 * r * (1.0f + scale[lane]);
  float y1 = x1 * r * (1.0f + scale[lane + 32]);
  float c0 = cosT[s * HDc + lane], c1 = cosT[s * HDc + lane + 32];
  float s0 = sinT[s * HDc + lane], s1 = sinT[s * HDc + lane + 32];
  float o0 = (y0 * c0 - y1 * s0) * postscale;
  float o1 = (y1 * c1 + y0 * s1) * postscale;
  __bf16* orow = out + (((size_t)b * nH + h) * Sc + s) * HDc;
  orow[lane]      = f2bf(o0);
  orow[lane + 32] = f2bf(o1);
}

__global__ __launch_bounds__(256)
void sigmoid_inplace(float* __restrict__ p, long n) {
  long i = (long)blockIdx.x * 256 + threadIdx.x;
  long stride = (long)gridDim.x * 256;
  for (; i < n; i += stride) p[i] = 1.0f / (1.0f + __expf(-p[i]));
}

// ---------------------------------------------------------------------------
// Flash attention (causal, GQA). grid = (S/64, H, B), block = 128 (4 waves).
// K tiles (key,d) and V^T tiles (d,key) async-copied into double-buffered LDS.
// One barrier per kv block; P staging is wave-private (no barrier needed).
// ---------------------------------------------------------------------------
__global__ __launch_bounds__(128)
void flash_attn(const __bf16* __restrict__ Qn, const __bf16* __restrict__ Kn,
                const __bf16* __restrict__ Vt, const float* __restrict__ gate,
                __bf16* __restrict__ ctxg) {
  __shared__ alignas(64) __bf16 sK[2][64][64];     // (key, d)  == B^T for scores
  __shared__ alignas(64) __bf16 sV[2][64][64];     // (d, key)  == B^T for context
  __shared__ alignas(64) __bf16 sP[4][16][64];     // per-wave probs (m, key)

  const int tid  = threadIdx.x;
  const int lane = tid & 31;
  const int wave = tid >> 5;
  const int hf   = lane >> 4;
  const int l15  = lane & 15;
  const int qb = blockIdx.x, h = blockIdx.y, b = blockIdx.z;
  const int g  = h >> 2;      // GS = H/G = 4

  const __bf16* Qbase = Qn + (((size_t)b * Hc + h) * Sc + qb * 64) * HDc;
  const __bf16* Kbase = Kn + (((size_t)b * Gc + g) * Sc) * HDc;       // (s, d)
  const __bf16* Vbase = Vt + (((size_t)b * Gc + g) * HDc) * Sc;       // (d, s)

  auto issue_kv = [&](int buf, int kb) {
#pragma unroll
    for (int i = 0; i < 4; ++i) {
      int idx = tid + i * 128;
      int r = idx >> 3;
      int c = (idx & 7) * 8;
      async_b128(&Kbase[((size_t)kb * 64 + r) * HDc + c], &sK[buf][r][c]);
      async_b128(&Vbase[(size_t)r * Sc + kb * 64 + c],    &sV[buf][r][c]);
    }
  };

  // persistent Q fragments: rows wave*16 + l15, two K-chunks over HD=64
  v16bf qa[2];
#pragma unroll
  for (int kci = 0; kci < 2; ++kci) {
    const __bf16* qrow = Qbase + (size_t)(wave * 16 + l15) * HDc + kci * 32 + hf * 16;
    v8bf alo = *(const v8bf*)(qrow);
    v8bf ahi = *(const v8bf*)(qrow + 8);
#pragma unroll
    for (int e = 0; e < 8; ++e) { qa[kci][e] = alo[e]; qa[kci][8 + e] = ahi[e]; }
  }

  const v8f vzero = {0.f, 0.f, 0.f, 0.f, 0.f, 0.f, 0.f, 0.f};
  v8f accO[4];
#pragma unroll
  for (int t = 0; t < 4; ++t) accO[t] = vzero;
  float mrow[8], lrow[8];
#pragma unroll
  for (int r = 0; r < 8; ++r) { mrow[r] = -__builtin_inff(); lrow[r] = 0.f; }

  issue_kv(0, 0);
  int buf = 0;
  for (int kb = 0; kb <= qb; ++kb) {
    wait_async0();
    __syncthreads();
    if (kb + 1 <= qb) issue_kv(buf ^ 1, kb + 1);   // overlap DMA with compute

    // ---- scores: S = Q x K^T (16 x 64 per wave) ----
    v8f sAcc[4];
#pragma unroll
    for (int t = 0; t < 4; ++t) sAcc[t] = vzero;
#pragma unroll
    for (int kc = 0; kc < 64; kc += 32) {
      v16bf a = qa[kc >> 5];
#pragma unroll
      for (int t = 0; t < 4; ++t) {
        v16bf bb = *(const v16bf*)&sK[buf][t * 16 + l15][kc + hf * 16];
        sAcc[t] = __builtin_amdgcn_wmma_f32_16x16x32_bf16(
            false, a, false, bb, (short)0, sAcc[t], false, false);
      }
    }

    // ---- causal mask + online softmax ----
    const bool diag = (kb == qb);
#pragma unroll
    for (int r = 0; r < 8; ++r) {
      int mrel = hf * 8 + r;
      int mabs = wave * 16 + mrel;                // row within 64-row q block
      float mx = -__builtin_inff();
#pragma unroll
      for (int t = 0; t < 4; ++t) {
        float v = sAcc[t][r];
        if (diag && (t * 16 + l15) > mabs) v = -__builtin_inff();
        sAcc[t][r] = v;
        mx = fmaxf(mx, v);
      }
      for (int m = 1; m < 16; m <<= 1) mx = fmaxf(mx, __shfl_xor(mx, m, 32));
      float nm = fmaxf(mrow[r], mx);
      float rsum = 0.f;
#pragma unroll
      for (int t = 0; t < 4; ++t) {
        float p = __expf(sAcc[t][r] - nm);
        sAcc[t][r] = p;
        rsum += p;
      }
      for (int m = 1; m < 16; m <<= 1) rsum += __shfl_xor(rsum, m, 32);
      float scl = __expf(mrow[r] - nm);
      lrow[r] = lrow[r] * scl + rsum;
      mrow[r] = nm;
#pragma unroll
      for (int t = 0; t < 4; ++t) accO[t][r] *= scl;
    }

    // stage P in LDS (wave-private region; wave-local DS ordering suffices)
#pragma unroll
    for (int t = 0; t < 4; ++t)
#pragma unroll
      for (int r = 0; r < 8; ++r)
        sP[wave][hf * 8 + r][t * 16 + l15] = f2bf(sAcc[t][r]);

    // ---- context: O += P x V ----
#pragma unroll
    for (int kc = 0; kc < 64; kc += 32) {
      const __bf16* prow = &sP[wave][l15][kc + hf * 16];
      v8bf alo = *(const v8bf*)(prow);
      v8bf ahi = *(const v8bf*)(prow + 8);
      v16bf a;
#pragma unroll
      for (int e = 0; e < 8; ++e) { a[e] = alo[e]; a[8 + e] = ahi[e]; }
#pragma unroll
      for (int t = 0; t < 4; ++t) {
        v16bf bb = *(const v16bf*)&sV[buf][t * 16 + l15][kc + hf * 16];
        accO[t] = __builtin_amdgcn_wmma_f32_16x16x32_bf16(
            false, a, false, bb, (short)0, accO[t], false, false);
      }
    }
    buf ^= 1;
  }

  // epilogue: normalize, gate, write bf16 (b, s, H*HD)
#pragma unroll
  for (int t = 0; t < 4; ++t)
#pragma unroll
    for (int r = 0; r < 8; ++r) {
      int mrel = hf * 8 + r;
      int srow = qb * 64 + wave * 16 + mrel;
      int d = t * 16 + l15;
      size_t off = ((size_t)(b * Sc + srow) * Hc + h) * HDc + d;
      float o = (accO[t][r] / lrow[r]) * gate[off];
      ctxg[off] = f2bf(o);
    }
}

// ---------------------------------------------------------------------------
// Host launcher
// ---------------------------------------------------------------------------
extern "C" void kernel_launch(void* const* d_in, const int* in_sizes, int n_in,
                              void* d_out, int out_size, void* d_ws, size_t ws_size,
                              hipStream_t stream) {
  (void)in_sizes; (void)n_in; (void)out_size; (void)ws_size;
  const float* x       = (const float*)d_in[0];
  const float* Wq      = (const float*)d_in[1];
  const float* Wk      = (const float*)d_in[2];
  const float* Wv      = (const float*)d_in[3];
  const float* Wg      = (const float*)d_in[4];
  const float* Wo      = (const float*)d_in[5];
  const float* q_scale = (const float*)d_in[6];
  const float* k_scale = (const float*)d_in[7];
  // d_in[8] = mask (causal, computed analytically)
  const float* cosT    = (const float*)d_in[9];
  const float* sinT    = (const float*)d_in[10];
  float* out = (float*)d_out;

  char* w = (char*)d_ws;
  __bf16* x_bf   = (__bf16*)(w + 0);            // 16,777,216 B
  __bf16* WqT_bf = (__bf16*)(w + 16777216);     //  8,388,608  (N,K)
  __bf16* WkT_bf = (__bf16*)(w + 25165824);     //  2,097,152
  __bf16* WvT_bf = (__bf16*)(w + 27262976);     //  2,097,152
  __bf16* WgT_bf = (__bf16*)(w + 29360128);     //  8,388,608
  __bf16* WoT_bf = (__bf16*)(w + 37748736);     //  8,388,608
  float*  QGf    = (float*) (w + 46137344);     // 33,554,432 (Q f32, later gate)
  float*  Kf     = (float*) (w + 79691776);     //  8,388,608
  float*  Vf     = (float*) (w + 88080384);     //  8,388,608
  __bf16* Qn     = (__bf16*)(w + 96468992);     // 16,777,216  (b,h,s,d)
  __bf16* Kn     = (__bf16*)(w + 113246208);    //  4,194,304  (b,g,s,d)
  __bf16* Vt     = (__bf16*)(w + 117440512);    //  4,194,304  (b,g,d,s)
  __bf16* Cg     = (__bf16*)(w + 79691776);     // 16,777,216 (reuses Kf+Vf)
  // total footprint: 121,634,816 B

  // 1) convert x; convert+transpose weights (one-time, so hot loops never transpose)
  cvt_f32_bf16<<<2048, 256, 0, stream>>>(x, x_bf, (long)BSc * Dc);
  cvt_transpose<<<dim3((Hc * HDc) / 64, Dc / 64), 256, 0, stream>>>(Wq, WqT_bf, Hc * HDc, Dc);
  cvt_transpose<<<dim3((Gc * HDc) / 64, Dc / 64), 256, 0, stream>>>(Wk, WkT_bf, Gc * HDc, Dc);
  cvt_transpose<<<dim3((Gc * HDc) / 64, Dc / 64), 256, 0, stream>>>(Wv, WvT_bf, Gc * HDc, Dc);
  cvt_transpose<<<dim3((Hc * HDc) / 64, Dc / 64), 256, 0, stream>>>(Wg, WgT_bf, Hc * HDc, Dc);
  cvt_transpose<<<dim3(Dc / 64, (Hc * HDc) / 64), 256, 0, stream>>>(Wo, WoT_bf, Dc, Hc * HDc);

  const float qscl = 0.125f;  // HD^-0.5 = 1/8

  // 2) Q projection + rmsnorm/rope epilogue
  gemm_bf16_wmma<<<dim3(Dc / 64, BSc / 64), 128, 0, stream>>>(
      x_bf, WqT_bf, QGf, BSc, Hc * HDc, Dc);
  qk_norm_rope<<<(Bc * Sc * Hc) / 8, 256, 0, stream>>>(
      QGf, Qn, q_scale, cosT, sinT, Hc, qscl, Bc * Sc * Hc);

  // 3) K projection + epilogue
  gemm_bf16_wmma<<<dim3((Gc * HDc) / 64, BSc / 64), 128, 0, stream>>>(
      x_bf, WkT_bf, Kf, BSc, Gc * HDc, Dc);
  qk_norm_rope<<<(Bc * Sc * Gc) / 8, 256, 0, stream>>>(
      Kf, Kn, k_scale, cosT, sinT, Gc, 1.0f, Bc * Sc * Gc);

  // 4) V projection + transpose to (b,g,d,s)
  gemm_bf16_wmma<<<dim3((Gc * HDc) / 64, BSc / 64), 128, 0, stream>>>(
      x_bf, WvT_bf, Vf, BSc, Gc * HDc, Dc);
  v_transpose<<<dim3(1, Sc / 64, Bc * Gc), 256, 0, stream>>>(Vf, Vt);

  // 5) gate projection (reuses Q f32 buffer) + sigmoid
  gemm_bf16_wmma<<<dim3(Dc / 64, BSc / 64), 128, 0, stream>>>(
      x_bf, WgT_bf, QGf, BSc, Hc * HDc, Dc);
  sigmoid_inplace<<<4096, 256, 0, stream>>>(QGf, (long)BSc * Hc * HDc);

  // 6) flash attention (causal, GQA), gate fused, bf16 output
  flash_attn<<<dim3(Sc / 64, Hc, Bc), 128, 0, stream>>>(Qn, Kn, Vt, QGf, Cg);

  // 7) output projection -> d_out (f32)
  gemm_bf16_wmma<<<dim3(Dc / 64, BSc / 64), 128, 0, stream>>>(
      Cg, WoT_bf, out, BSc, Dc, Hc * HDc);
}
